// GatedSpikingReservoirStep_10806137717143
// MI455X (gfx1250) — matrix operand order repaired and checked by reading, not compiled, not run on hardware
//
#include <hip/hip_runtime.h>

typedef __attribute__((ext_vector_type(16))) __bf16 v16bf;
typedef __attribute__((ext_vector_type(8)))  __bf16 v8bf;
typedef __attribute__((ext_vector_type(4)))  __bf16 v4bf;
typedef __attribute__((ext_vector_type(8)))  float  v8f;

#define B_DIM 4096
#define I_DIM 512
#define D_DIM 2048
#define MD    2560
#define BM    64      // block tile rows (batch)
#define BN    64      // block tile cols (D)
#define KT    64      // K tile
#define LDSTR 72      // padded LDS stride in bf16 elements (144B, 16B aligned)

// ---- shared memory arena (pass-1 tiles and pass-2 double buffers overlay) ----
// pass 1: A1 (64x72 bf16 = 9216B) | B1[4] (64x72 bf16 = 9216B each) -> 46080B
// pass 2: buf[k] = { A (9216B), B (9216B) } at k*18432 -> 36864B
#define SMEM_BYTES 46080
#define P1A_OFF    0
#define P1B_OFF(j) (9216 + (j) * 9216)
#define P2A_OFF(k) ((k) * 18432)
#define P2B_OFF(k) ((k) * 18432 + 9216)

__device__ __forceinline__ void cvt_store4(__bf16* dst, float4 v) {
    v4bf o;
    o[0] = (__bf16)v.x; o[1] = (__bf16)v.y; o[2] = (__bf16)v.z; o[3] = (__bf16)v.w;
    *(v4bf*)dst = o;
}

// A fragment (16x32 bf16): elems 0..7 = K[h*8..h*8+7], elems 8..15 = K[16+h*8..].
__device__ __forceinline__ v16bf ld_frag_a(const __bf16* base) {
    v8bf lo = *(const v8bf*)(base);
    v8bf hi = *(const v8bf*)(base + 16);
    return __builtin_shufflevector(lo, hi, 0,1,2,3,4,5,6,7,8,9,10,11,12,13,14,15);
}

// B fragment (32x16 bf16): lane = N, elems 0..7 = K[h*8..h*8+7], elems 8..15 = K[16+h*8..].
__device__ __forceinline__ v16bf ld_frag_b(const __bf16* base) {
    v8bf lo = *(const v8bf*)(base);
    v8bf hi = *(const v8bf*)(base + 16);
    return __builtin_shufflevector(lo, hi, 0,1,2,3,4,5,6,7,8,9,10,11,12,13,14,15);
}

__device__ __forceinline__ float sigmoidf(float x) {
    return 1.0f / (1.0f + expf(-x));
}

__global__ __launch_bounds__(256)
void reservoir_step_kernel(const float* __restrict__ inputs,
                           const float* __restrict__ state,
                           const float* __restrict__ W_in,
                           const float* __restrict__ W_res,
                           const float* __restrict__ W_gate,
                           float* __restrict__ out) {
    __shared__ __align__(16) unsigned char smem[SMEM_BYTES];

    const int tid  = threadIdx.x;
    const int lane = tid & 31;
    const int wv   = tid >> 5;      // 0..7
    const int wm   = wv >> 2;       // 0..1  (32-row strip within 64)
    const int wn   = wv & 3;        // 0..3  (16-col strip within 64)
    const int h    = lane >> 4;     // lane half
    const int nl   = lane & 15;

    const int bn0 = blockIdx.x * BN;   // D-column base
    const int bm0 = blockIdx.y * BM;   // batch-row base

    // acc[group][msub]; group 0: input+reservoir, 1: i, 2: f, 3: o
    v8f acc[4][2];
    #pragma unroll
    for (int j = 0; j < 4; ++j)
        #pragma unroll
        for (int ms = 0; ms < 2; ++ms)
            #pragma unroll
            for (int e = 0; e < 8; ++e) acc[j][ms][e] = 0.0f;

    // ================= Pass 1: K over I_DIM (inputs / W_in + 3 gate slabs) ====
    {
        __bf16* sA = (__bf16*)(smem + P1A_OFF);
        const float* wbs[4] = {
            W_in   + (size_t)bn0 * I_DIM,
            W_gate + (size_t)bn0 * I_DIM,
            W_gate + (size_t)(bn0 +     D_DIM) * I_DIM,
            W_gate + (size_t)(bn0 + 2 * D_DIM) * I_DIM };

        for (int kt = 0; kt < I_DIM / KT; ++kt) {
            const int k0 = kt * KT;
            // ---- issue ALL tile loads into distinct registers first --------
            float4 ra[4], rb[4][4];
            #pragma unroll
            for (int r = 0; r < 4; ++r) {
                int s = tid + 256 * r;
                int row = s >> 4, kk = (s & 15) * 4;
                ra[r] = *(const float4*)(inputs + (size_t)(bm0 + row) * I_DIM + k0 + kk);
            }
            #pragma unroll
            for (int j = 0; j < 4; ++j)
                #pragma unroll
                for (int r = 0; r < 4; ++r) {
                    int s = tid + 256 * r;
                    int row = s >> 4, kk = (s & 15) * 4;
                    rb[j][r] = *(const float4*)(wbs[j] + (size_t)row * I_DIM + k0 + kk);
                }
            if (kt + 1 < I_DIM / KT)
                __builtin_prefetch(inputs + (size_t)(bm0 + (tid >> 4)) * I_DIM + k0 + KT, 0, 0);
            // ---- then convert + store (one wait, batched ds_stores) --------
            #pragma unroll
            for (int r = 0; r < 4; ++r) {
                int s = tid + 256 * r;
                int row = s >> 4, kk = (s & 15) * 4;
                cvt_store4(&sA[row * LDSTR + kk], ra[r]);
            }
            #pragma unroll
            for (int j = 0; j < 4; ++j) {
                __bf16* sB = (__bf16*)(smem + P1B_OFF(j));
                #pragma unroll
                for (int r = 0; r < 4; ++r) {
                    int s = tid + 256 * r;
                    int row = s >> 4, kk = (s & 15) * 4;
                    cvt_store4(&sB[row * LDSTR + kk], rb[j][r]);
                }
            }
            __syncthreads();
            // ---- compute: 2 A-frags + 4 B-frags -> 8 WMMAs per k-step ------
            #pragma unroll
            for (int ks = 0; ks < KT; ks += 32) {
                v16bf a[2];
                #pragma unroll
                for (int ms = 0; ms < 2; ++ms)
                    a[ms] = ld_frag_a(&sA[(wm * 32 + ms * 16 + nl) * LDSTR + ks + h * 8]);
                v16bf bfr[4];
                #pragma unroll
                for (int j = 0; j < 4; ++j)
                    bfr[j] = ld_frag_b((const __bf16*)(smem + P1B_OFF(j)) +
                                       (wn * 16 + nl) * LDSTR + ks + h * 8);
                #pragma unroll
                for (int j = 0; j < 4; ++j)
                    #pragma unroll
                    for (int ms = 0; ms < 2; ++ms)
                        acc[j][ms] = __builtin_amdgcn_wmma_f32_16x16x32_bf16(
                            false, a[ms], false, bfr[j], (short)0, acc[j][ms], false, false);
            }
            __syncthreads();
        }
    }

    // ================= Pass 2: K over D_DIM (prev_state / W_res) =============
    // Double-buffered in LDS: stage kt+1 into registers while computing kt.
    {
        const int NT2 = D_DIM / KT;   // 32
        float4 ra[4], rb[4];
        // prologue: load tile 0 and store into buffer 0
        {
            #pragma unroll
            for (int r = 0; r < 4; ++r) {
                int s = tid + 256 * r;
                int row = s >> 4, kk = (s & 15) * 4;
                ra[r] = *(const float4*)(state + (size_t)(bm0 + row) * MD + kk);
                rb[r] = *(const float4*)(W_res + (size_t)(bn0 + row) * D_DIM + kk);
            }
            __bf16* sA = (__bf16*)(smem + P2A_OFF(0));
            __bf16* sB = (__bf16*)(smem + P2B_OFF(0));
            #pragma unroll
            for (int r = 0; r < 4; ++r) {
                int s = tid + 256 * r;
                int row = s >> 4, kk = (s & 15) * 4;
                cvt_store4(&sA[row * LDSTR + kk], ra[r]);
                cvt_store4(&sB[row * LDSTR + kk], rb[r]);
            }
            __syncthreads();
        }
        for (int kt = 0; kt < NT2; ++kt) {
            const int cur = kt & 1;
            // issue next tile's global loads (no wait until cvt below)
            if (kt + 1 < NT2) {
                const int k1 = (kt + 1) * KT;
                #pragma unroll
                for (int r = 0; r < 4; ++r) {
                    int s = tid + 256 * r;
                    int row = s >> 4, kk = (s & 15) * 4;
                    ra[r] = *(const float4*)(state + (size_t)(bm0 + row) * MD + k1 + kk);
                    rb[r] = *(const float4*)(W_res + (size_t)(bn0 + row) * D_DIM + k1 + kk);
                }
            }
            // compute on current buffer: 4 A-frags + 2 B-frags -> 4 WMMAs
            {
                const __bf16* sA = (const __bf16*)(smem + P2A_OFF(cur));
                const __bf16* sB = (const __bf16*)(smem + P2B_OFF(cur));
                #pragma unroll
                for (int ks = 0; ks < KT; ks += 32) {
                    v16bf b = ld_frag_b(&sB[(wn * 16 + nl) * LDSTR + ks + h * 8]);
                    #pragma unroll
                    for (int ms = 0; ms < 2; ++ms) {
                        v16bf a = ld_frag_a(&sA[(wm * 32 + ms * 16 + nl) * LDSTR + ks + h * 8]);
                        acc[0][ms] = __builtin_amdgcn_wmma_f32_16x16x32_bf16(
                            false, a, false, b, (short)0, acc[0][ms], false, false);
                    }
                }
            }
            // store staged registers into the other buffer
            if (kt + 1 < NT2) {
                __bf16* sA = (__bf16*)(smem + P2A_OFF(cur ^ 1));
                __bf16* sB = (__bf16*)(smem + P2B_OFF(cur ^ 1));
                #pragma unroll
                for (int r = 0; r < 4; ++r) {
                    int s = tid + 256 * r;
                    int row = s >> 4, kk = (s & 15) * 4;
                    cvt_store4(&sA[row * LDSTR + kk], ra[r]);
                    cvt_store4(&sB[row * LDSTR + kk], rb[r]);
                }
            }
            __syncthreads();
        }
    }

    // ================= Epilogue: gates, leak, tanh, spike =====================
    const int colg = bn0 + wn * 16 + nl;           // < 2048 always
    #pragma unroll
    for (int ms = 0; ms < 2; ++ms) {
        #pragma unroll
        for (int v = 0; v < 8; ++v) {
            const int rowg = bm0 + wm * 32 + ms * 16 + v + 8 * h;
            const float prev = state[(size_t)rowg * MD + colg];
            const float ig = sigmoidf(acc[1][ms][v]);
            const float fg = sigmoidf(acc[2][ms][v]);
            const float og = sigmoidf(acc[3][ms][v]);
            float ns = 0.8f * (fg * prev) + 0.2f * tanhf(ig * acc[0][ms][v]);
            ns *= og;
            ns = (ns > 0.5f) ? (ns - 0.5f) : ns;
            out[(size_t)rowg * MD + colg] = ns;
        }
    }
}

// Zero the padded columns out[:, 2048:2560] (4096 x 512 floats).
__global__ __launch_bounds__(256)
void pad_zero_kernel(float* __restrict__ out) {
    const int t   = blockIdx.x * 256 + threadIdx.x;   // 0 .. 524287
    const int row = t >> 7;                           // 128 float4 per row
    const int c4  = t & 127;
    float4 z; z.x = 0.f; z.y = 0.f; z.z = 0.f; z.w = 0.f;
    *(float4*)(out + (size_t)row * MD + D_DIM + c4 * 4) = z;
}

extern "C" void kernel_launch(void* const* d_in, const int* in_sizes, int n_in,
                              void* d_out, int out_size, void* d_ws, size_t ws_size,
                              hipStream_t stream) {
    (void)in_sizes; (void)n_in; (void)out_size; (void)d_ws; (void)ws_size;
    const float* inputs = (const float*)d_in[0];   // (4096, 512)
    const float* state  = (const float*)d_in[1];   // (4096, 2560)
    const float* W_in   = (const float*)d_in[2];   // (2048, 512)
    const float* W_res  = (const float*)d_in[3];   // (2048, 2048)
    const float* W_gate = (const float*)d_in[4];   // (6144, 512)
    float* out = (float*)d_out;                    // (4096, 2560)

    pad_zero_kernel<<<(B_DIM * (MD - D_DIM) / 4) / 256, 256, 0, stream>>>(out);

    dim3 grid(D_DIM / BN, B_DIM / BM);             // (32, 64)
    reservoir_step_kernel<<<grid, 256, 0, stream>>>(inputs, state, W_in, W_res,
                                                    W_gate, out);
}
